// LieMultiHeadAttention_77524159693026
// MI455X (gfx1250) — matrix operand order repaired and checked by reading, not compiled, not run on hardware
//
#include <hip/hip_runtime.h>
#include <hip/hip_fp16.h>

typedef __attribute__((ext_vector_type(16))) _Float16 v16h;
typedef __attribute__((ext_vector_type(8)))  _Float16 v8h;
typedef __attribute__((ext_vector_type(8)))  float    v8f;
typedef unsigned int u32x4 __attribute__((ext_vector_type(4)));
typedef int          i32x4 __attribute__((ext_vector_type(4)));
typedef int          i32x8 __attribute__((ext_vector_type(8)));

#define BB 4
#define NN 2048
#define CC 256
#define HH 4
#define DD 64
#define OO3 (3*CC)

// ---- helpers -------------------------------------------------------------

static __device__ __forceinline__ v16h cat8(v8h lo, v8h hi) {
  v16h r;
#pragma unroll
  for (int i = 0; i < 8; ++i) { r[i] = lo[i]; r[i + 8] = hi[i]; }
  return r;
}

static __device__ __forceinline__ v8f wmma_f16(v16h a, v16h b, v8f c) {
  return __builtin_amdgcn_wmma_f32_16x16x32_f16(
      /*neg_a=*/false, a, /*neg_b=*/false, b,
      /*c_mod=*/(short)0, c, /*reuse_a=*/false, /*reuse_b=*/false);
}

static __device__ __forceinline__ float redmax16(float v) {
#pragma unroll
  for (int m = 1; m < 16; m <<= 1) v = fmaxf(v, __shfl_xor(v, m, 16));
  return v;
}
static __device__ __forceinline__ float redsum16(float v) {
#pragma unroll
  for (int m = 1; m < 16; m <<= 1) v += __shfl_xor(v, m, 16);
  return v;
}

// Tensor Data Mover: 2D tile load (2-byte elements) global -> LDS.
// D# layout per CDNA5 ISA 8.3/8.4: group0 = {flags, lds_addr, global_addr,
// type=2}; group1 = {data_size, tensor dims, tile dims, dim0 stride}.
// tensor_dim* set equal to the tile so every element is in-bounds
// (global_addr already points at the tile start). This toolchain's builtin
// takes 6 args: (g0, g1, g2, g3, g_extra, cpol).
static __device__ __forceinline__ void tdm_load_tile_2d(
    unsigned int lds_addr, const void* gptr,
    unsigned int tile0, unsigned int tile1, unsigned long long stride0) {
  unsigned long long ga = (unsigned long long)(uintptr_t)gptr;
  u32x4 g0;
  g0.x = 1u;                                  // count=1, user descriptor
  g0.y = lds_addr;                            // LDS byte address
  g0.z = (unsigned int)ga;                    // global_addr[31:0]
  g0.w = (unsigned int)((ga >> 32) & 0x01ffffffu) | (2u << 30);  // [56:32]|type=2
  i32x8 g1;
  g1[0] = (1 << 16);                          // data_size=1 -> 2 bytes
  g1[1] = (int)((tile0 & 0xffffu) << 16);     // tensor_dim0[15:0]
  g1[2] = (int)((tile0 >> 16) & 0xffffu) |    // tensor_dim0[31:16]
          (int)((tile1 & 0xffffu) << 16);     // tensor_dim1[15:0]
  g1[3] = (int)((tile1 >> 16) & 0xffffu) |    // tensor_dim1[31:16]
          (int)((tile0 & 0xffffu) << 16);     // tile_dim0
  g1[4] = (int)(tile1 & 0xffffu);             // tile_dim1 (tile_dim2=0)
  g1[5] = (int)(stride0 & 0xffffffffu);       // tensor_dim0_stride[31:0]
  g1[6] = (int)((stride0 >> 32) & 0xffffu);   // stride[47:32], dim1_stride=0
  g1[7] = 0;
  i32x4 z4 = {0, 0, 0, 0};
  i32x8 z8 = {0, 0, 0, 0, 0, 0, 0, 0};
  __builtin_amdgcn_tensor_load_to_lds(g0, g1, z4, z4, z8, 0);
}

// ---- stage 0: fp32 -> fp16 conversion ------------------------------------

__global__ void k_f32_to_f16(const float* __restrict__ in,
                             _Float16* __restrict__ out, int n) {
  int i = blockIdx.x * blockDim.x + threadIdx.x;
  if (i < n) out[i] = (_Float16)in[i];
}

// ---- stage 0b: RP[b,j,c] = pos_scale * sum_d R[b,c,d] * P[j,d] -----------

__global__ void k_rp(const float* __restrict__ R, const float* __restrict__ P,
                     const float* __restrict__ pos_scale,
                     float* __restrict__ RP) {
  int idx = blockIdx.x * blockDim.x + threadIdx.x;  // b*NN + j
  if (idx >= BB * NN) return;
  int b = idx / NN, j = idx % NN;
  float p0 = P[j * 3 + 0], p1 = P[j * 3 + 1], p2 = P[j * 3 + 2];
  float ps = pos_scale[0];
  const float* Rb = R + b * 9;
#pragma unroll
  for (int c = 0; c < 3; ++c)
    RP[idx * 3 + c] = ps * (Rb[c * 3 + 0] * p0 + Rb[c * 3 + 1] * p1 + Rb[c * 3 + 2] * p2);
}

// ---- stage 1: QKV GEMM (x @ qkv_w.T + qkv_b), scatter to q/k/vT ----------

__global__ void __launch_bounds__(256) k_qkv(
    const _Float16* __restrict__ xh, const _Float16* __restrict__ wh,
    const float* __restrict__ bias, _Float16* __restrict__ q,
    _Float16* __restrict__ k, _Float16* __restrict__ vT) {
  const int lane = threadIdx.x & 31;
  const int w = (blockIdx.x * blockDim.x + threadIdx.x) >> 5;
  const int ct = w % (OO3 / 64);
  const int rt = w / (OO3 / 64);
  const int n0 = rt * 16;
  const int o0 = ct * 64;
  const int l15 = lane & 15;
  const int hi16 = (lane >> 4) & 1;

  v8f acc0 = {}, acc1 = {}, acc2 = {}, acc3 = {};
  const _Float16* xrow = xh + (size_t)(n0 + l15) * CC;
#pragma unroll 2
  for (int c0 = 0; c0 < CC; c0 += 32) {
    v16h a = cat8(*(const v8h*)(xrow + c0 + hi16 * 16),
                  *(const v8h*)(xrow + c0 + hi16 * 16 + 8));
    const _Float16* wb = wh + c0 + hi16 * 16;
    v16h b0 = *(const v16h*)(wb + (size_t)(o0 + 0 + l15) * CC);
    v16h b1 = *(const v16h*)(wb + (size_t)(o0 + 16 + l15) * CC);
    v16h b2 = *(const v16h*)(wb + (size_t)(o0 + 32 + l15) * CC);
    v16h b3 = *(const v16h*)(wb + (size_t)(o0 + 48 + l15) * CC);
    acc0 = wmma_f16(a, b0, acc0);
    acc1 = wmma_f16(a, b1, acc1);
    acc2 = wmma_f16(a, b2, acc2);
    acc3 = wmma_f16(a, b3, acc3);
  }

  const int tsel = o0 >> 8;      // 0=q 1=k 2=v, uniform per wave
  const int h = (o0 >> 6) & 3;   // head, uniform per wave
  v8f accs[4] = {acc0, acc1, acc2, acc3};
#pragma unroll
  for (int t = 0; t < 4; ++t) {
    int o = o0 + t * 16 + l15;
    float bv = bias[o];
    int d = o & 63;
#pragma unroll
    for (int r = 0; r < 8; ++r) {
      int ng = n0 + r + hi16 * 8;
      int bb = ng >> 11;
      int nn = ng & (NN - 1);
      float val = accs[t][r] + bv;
      if (tsel == 0)
        q[(((size_t)bb * HH + h) * NN + nn) * DD + d] = (_Float16)(val * 0.125f);
      else if (tsel == 1)
        k[(((size_t)bb * HH + h) * NN + nn) * DD + d] = (_Float16)val;
      else
        vT[(((size_t)bb * HH + h) * DD + d) * NN + nn] = (_Float16)val;
    }
  }
}

// ---- stage 2: flash attention with TDM double-buffered K/V staging -------
// Block = 8 waves = 8 consecutive 16-row i-tiles of one (b,h). All waves walk
// the same 32-key chunks; wave 0 DMAs each chunk's K (4KB contiguous) and Vt
// (64 rows x 64B, stride N) tiles into LDS via the Tensor Data Mover,
// double-buffered, synchronized with s_wait_tensorcnt + block barrier.

__global__ void __launch_bounds__(256) k_attn(
    const _Float16* __restrict__ q, const _Float16* __restrict__ k,
    const _Float16* __restrict__ vT, const float* __restrict__ RP,
    const float* __restrict__ P, _Float16* __restrict__ outh) {
  __shared__ __align__(128) _Float16 sK[2][32 * DD];   // [j 0..31][d 0..63]
  __shared__ __align__(128) _Float16 sV[2][DD * 32];   // [d 0..63][j 0..31]
  __shared__ __align__(32)  _Float16 sP[8][16 * 32];

  const int lane = threadIdx.x & 31;
  const int wg = threadIdx.x >> 5;
  const int bh = blockIdx.x >> 4;                       // 16 blocks per (b,h)
  const int i0 = (((blockIdx.x & 15) << 3) + wg) * 16;  // this wave's rows
  const int b = bh / HH, h = bh % HH;
  const int l15 = lane & 15, hi16 = (lane >> 4) & 1;

  const _Float16* Q = q + (size_t)bh * NN * DD;
  const _Float16* K = k + (size_t)bh * NN * DD;
  const _Float16* Vt = vT + (size_t)bh * DD * NN;

  const _Float16* qrow = Q + (size_t)(i0 + l15) * DD;
  v16h qa0 = cat8(*(const v8h*)(qrow + hi16 * 16),
                  *(const v8h*)(qrow + hi16 * 16 + 8));
  v16h qa1 = cat8(*(const v8h*)(qrow + 32 + hi16 * 16),
                  *(const v8h*)(qrow + 32 + hi16 * 16 + 8));

  float rpx[8], rpy[8], rpz[8];
  const float* RPb = RP + (size_t)b * NN * 3;
#pragma unroll
  for (int r = 0; r < 8; ++r) {
    int i = i0 + r + hi16 * 8;
    rpx[r] = RPb[i * 3 + 0];
    rpy[r] = RPb[i * 3 + 1];
    rpz[r] = RPb[i * 3 + 2];
  }

  v8f o0a = {}, o1a = {}, o2a = {}, o3a = {};
  float m[8], l[8];
#pragma unroll
  for (int r = 0; r < 8; ++r) { m[r] = -3.0e38f; l[r] = 0.f; }

  const int NCHUNK = NN / 32;
  // prologue: stage chunk 0 into buffer 0
  if (wg == 0) {
    tdm_load_tile_2d((unsigned int)(uintptr_t)&sK[0][0], K, 32 * DD, 1, 32 * DD);
    tdm_load_tile_2d((unsigned int)(uintptr_t)&sV[0][0], Vt, 32, DD, NN);
    __builtin_amdgcn_s_wait_tensorcnt((short)0);
  }
  __syncthreads();

  for (int jc = 0; jc < NCHUNK; ++jc) {
    const int j0 = jc * 32;
    const int buf = jc & 1;
    if (wg == 0 && jc + 1 < NCHUNK) {  // prefetch next chunk into other buffer
      tdm_load_tile_2d((unsigned int)(uintptr_t)&sK[buf ^ 1][0],
                       K + (size_t)(j0 + 32) * DD, 32 * DD, 1, 32 * DD);
      tdm_load_tile_2d((unsigned int)(uintptr_t)&sV[buf ^ 1][0],
                       Vt + (j0 + 32), 32, DD, NN);
    }
    const _Float16* Kb = &sK[buf][0];
    const _Float16* Vb = &sV[buf][0];

    // S = Q K^T for two 16x16 tiles (32 keys), K fragments from LDS
    const _Float16* krow0 = Kb + (size_t)l15 * DD + hi16 * 16;
    const _Float16* krow1 = Kb + (size_t)(16 + l15) * DD + hi16 * 16;
    v16h kb00 = *(const v16h*)(krow0);
    v16h kb01 = *(const v16h*)(krow0 + 32);
    v16h kb10 = *(const v16h*)(krow1);
    v16h kb11 = *(const v16h*)(krow1 + 32);
    v8f s0 = {}, s1 = {};
    s0 = wmma_f16(qa0, kb00, s0);
    s0 = wmma_f16(qa1, kb01, s0);
    s1 = wmma_f16(qa0, kb10, s1);
    s1 = wmma_f16(qa1, kb11, s1);

    // positional bias + online softmax update
    int ja = (j0 + l15) * 3, jb = (j0 + 16 + l15) * 3;
    float pa0 = P[ja], pa1 = P[ja + 1], pa2 = P[ja + 2];
    float pb0 = P[jb], pb1 = P[jb + 1], pb2 = P[jb + 2];
#pragma unroll
    for (int r = 0; r < 8; ++r) {
      float sv0 = s0[r] + rpx[r] * pa0 + rpy[r] * pa1 + rpz[r] * pa2;
      float sv1 = s1[r] + rpx[r] * pb0 + rpy[r] * pb1 + rpz[r] * pb2;
      float mnew = fmaxf(m[r], redmax16(fmaxf(sv0, sv1)));
      float alpha = __expf(m[r] - mnew);
      m[r] = mnew;
      float e0 = __expf(sv0 - mnew);
      float e1 = __expf(sv1 - mnew);
      l[r] = l[r] * alpha + redsum16(e0 + e1);
      o0a[r] *= alpha; o1a[r] *= alpha; o2a[r] *= alpha; o3a[r] *= alpha;
      int ii = r + hi16 * 8;
      sP[wg][ii * 32 + l15] = (_Float16)e0;
      sP[wg][ii * 32 + 16 + l15] = (_Float16)e1;
    }
    // reload probabilities in A-fragment layout (wave-private LDS tile)
    v16h pfrag = cat8(*(const v8h*)&sP[wg][l15 * 32 + hi16 * 16],
                      *(const v8h*)&sP[wg][l15 * 32 + hi16 * 16 + 8]);
    // O += P @ V, V fragments from LDS (row d, contiguous j)
    const _Float16* vbase = Vb + hi16 * 16;
    v16h vb0 = *(const v16h*)(vbase + (size_t)(0 * 16 + l15) * 32);
    v16h vb1 = *(const v16h*)(vbase + (size_t)(1 * 16 + l15) * 32);
    v16h vb2 = *(const v16h*)(vbase + (size_t)(2 * 16 + l15) * 32);
    v16h vb3 = *(const v16h*)(vbase + (size_t)(3 * 16 + l15) * 32);
    o0a = wmma_f16(pfrag, vb0, o0a);
    o1a = wmma_f16(pfrag, vb1, o1a);
    o2a = wmma_f16(pfrag, vb2, o2a);
    o3a = wmma_f16(pfrag, vb3, o3a);

    // wave 0: next chunk's DMA complete; barrier releases all waves and also
    // guarantees everyone is done reading `buf` before it is refilled.
    if (wg == 0) __builtin_amdgcn_s_wait_tensorcnt((short)0);
    __syncthreads();
  }

  // normalize and store (B,N,C) f16 for the projection GEMM
#pragma unroll
  for (int r = 0; r < 8; ++r) {
    float inv = 1.0f / l[r];
    int i = i0 + r + hi16 * 8;
    size_t base = ((size_t)b * NN + i) * CC + h * DD;
    outh[base + 0 * 16 + l15] = (_Float16)(o0a[r] * inv);
    outh[base + 1 * 16 + l15] = (_Float16)(o1a[r] * inv);
    outh[base + 2 * 16 + l15] = (_Float16)(o2a[r] * inv);
    outh[base + 3 * 16 + l15] = (_Float16)(o3a[r] * inv);
  }
}

// ---- stage 3: output projection (attn @ proj_w.T + proj_b), fp32 out -----

__global__ void __launch_bounds__(256) k_proj(
    const _Float16* __restrict__ ah, const _Float16* __restrict__ wh,
    const float* __restrict__ bias, float* __restrict__ out) {
  const int lane = threadIdx.x & 31;
  const int w = (blockIdx.x * blockDim.x + threadIdx.x) >> 5;
  const int ct = w % (CC / 64);
  const int rt = w / (CC / 64);
  const int n0 = rt * 16, o0 = ct * 64;
  const int l15 = lane & 15, hi16 = (lane >> 4) & 1;

  v8f acc0 = {}, acc1 = {}, acc2 = {}, acc3 = {};
  const _Float16* arow = ah + (size_t)(n0 + l15) * CC;
#pragma unroll 2
  for (int c0 = 0; c0 < CC; c0 += 32) {
    v16h a = cat8(*(const v8h*)(arow + c0 + hi16 * 16),
                  *(const v8h*)(arow + c0 + hi16 * 16 + 8));
    const _Float16* wb = wh + c0 + hi16 * 16;
    v16h b0 = *(const v16h*)(wb + (size_t)(o0 + 0 + l15) * CC);
    v16h b1 = *(const v16h*)(wb + (size_t)(o0 + 16 + l15) * CC);
    v16h b2 = *(const v16h*)(wb + (size_t)(o0 + 32 + l15) * CC);
    v16h b3 = *(const v16h*)(wb + (size_t)(o0 + 48 + l15) * CC);
    acc0 = wmma_f16(a, b0, acc0);
    acc1 = wmma_f16(a, b1, acc1);
    acc2 = wmma_f16(a, b2, acc2);
    acc3 = wmma_f16(a, b3, acc3);
  }
  v8f accs[4] = {acc0, acc1, acc2, acc3};
#pragma unroll
  for (int t = 0; t < 4; ++t) {
    int o = o0 + t * 16 + l15;
    float bv = bias[o];
#pragma unroll
    for (int r = 0; r < 8; ++r)
      out[(size_t)(n0 + r + hi16 * 8) * CC + o] = accs[t][r] + bv;
  }
}

// ---- host launcher -------------------------------------------------------

extern "C" void kernel_launch(void* const* d_in, const int* in_sizes, int n_in,
                              void* d_out, int out_size, void* d_ws,
                              size_t ws_size, hipStream_t stream) {
  (void)in_sizes; (void)n_in; (void)out_size; (void)ws_size;
  const float* x = (const float*)d_in[0];
  const float* R = (const float*)d_in[1];
  const float* P = (const float*)d_in[2];
  const float* qkv_w = (const float*)d_in[3];
  const float* qkv_b = (const float*)d_in[4];
  const float* proj_w = (const float*)d_in[5];
  const float* proj_b = (const float*)d_in[6];
  const float* pos_scale = (const float*)d_in[7];
  float* out = (float*)d_out;

  char* p = (char*)d_ws;
  _Float16* xh = (_Float16*)p;  p += (size_t)BB * NN * CC * 2;
  _Float16* qwh = (_Float16*)p; p += (size_t)OO3 * CC * 2;
  _Float16* pwh = (_Float16*)p; p += (size_t)CC * CC * 2;
  float* RPws = (float*)p;      p += (size_t)BB * NN * 3 * 4;
  _Float16* qws = (_Float16*)p; p += (size_t)BB * HH * NN * DD * 2;
  _Float16* kws = (_Float16*)p; p += (size_t)BB * HH * NN * DD * 2;
  _Float16* vtws = (_Float16*)p; p += (size_t)BB * HH * DD * NN * 2;
  _Float16* aoh = (_Float16*)p; p += (size_t)BB * NN * CC * 2;

  int nx = BB * NN * CC;
  k_f32_to_f16<<<(nx + 255) / 256, 256, 0, stream>>>(x, xh, nx);
  int nw = OO3 * CC;
  k_f32_to_f16<<<(nw + 255) / 256, 256, 0, stream>>>(qkv_w, qwh, nw);
  int np = CC * CC;
  k_f32_to_f16<<<(np + 255) / 256, 256, 0, stream>>>(proj_w, pwh, np);
  k_rp<<<(BB * NN + 255) / 256, 256, 0, stream>>>(R, P, pos_scale, RPws);

  k_qkv<<<(BB * NN / 16) * (OO3 / 64) / 8, 256, 0, stream>>>(
      xh, qwh, qkv_b, qws, kws, vtws);
  k_attn<<<(BB * HH * (NN / 16)) / 8, 256, 0, stream>>>(
      qws, kws, vtws, RPws, P, aoh);
  k_proj<<<(BB * NN / 16) * (CC / 64) / 8, 256, 0, stream>>>(
      aoh, pwh, proj_b, out);
}
